// Regression_8735963480771
// MI455X (gfx1250) — matrix operand order repaired and checked by reading, not compiled, not run on hardware
//
#include <hip/hip_runtime.h>
#include <hip/hip_bf16.h>
#include <math.h>

typedef __bf16 bf16x16 __attribute__((ext_vector_type(16)));
typedef __bf16 bf16x8  __attribute__((ext_vector_type(8)));
typedef float  f32x8   __attribute__((ext_vector_type(8)));

#define WMMA_BF16(a, b, c) \
  __builtin_amdgcn_wmma_f32_16x16x32_bf16(false, (a), false, (b), (short)0, (c), false, false)

// Join two 8-element bf16 vector loads into one 16-element WMMA operand.
__device__ __forceinline__ bf16x16 join16(const __bf16* p0, const __bf16* p1) {
  bf16x8 lo = *(const bf16x8*)p0;
  bf16x8 hi = *(const bf16x8*)p1;
  return __builtin_shufflevector(lo, hi, 0, 1, 2, 3, 4, 5, 6, 7,
                                         8, 9, 10, 11, 12, 13, 14, 15);
}

// ---------------------------------------------------------------------------
// Pack f32 [rows, K] (row stride ld_src) -> zero-padded bf16 [rows, Kp].
// ---------------------------------------------------------------------------
__global__ void pack_bf16(const float* __restrict__ src, int ld_src,
                          __bf16* __restrict__ dst, int rows, int K, int Kp) {
  const int i = blockIdx.x * blockDim.x + threadIdx.x;
  if (i < rows * Kp) {
    const int r = i / Kp, k = i % Kp;
    dst[i] = (__bf16)((k < K) ? src[(size_t)r * ld_src + k] : 0.f);
  }
}

// ---------------------------------------------------------------------------
// GEMM on packed operands: C[m,n] = sum_k Xp[m,k]*Wp[n,k] (+b0[n]+b1[n]).
// Requires M%16==0, N%16==0, Kp%32==0. One wave per 16x16 tile.
// A-frag: lane m=l%16, 16B runs at kb+half*8 and kb+16+half*8.
// B-frag: lane n=l%16, 16B runs at kb+half*8 and kb+16+half*8.
// ---------------------------------------------------------------------------
__global__ void __launch_bounds__(32)
gemm_bf16_wmma(const __bf16* __restrict__ Xp, const __bf16* __restrict__ Wp,
               int Kp,
               const float* __restrict__ b0, const float* __restrict__ b1,
               float* __restrict__ C, int ldc, int N) {
  const int lane = threadIdx.x & 31;
  const int half = lane >> 4;
  const int l16  = lane & 15;
  const int mA = blockIdx.y * 16 + l16;
  const int nB = blockIdx.x * 16 + l16;

  const __bf16* arow = Xp + (size_t)mA * Kp + (half << 3);
  const __bf16* brow = Wp + (size_t)nB * Kp + (half << 3);

  f32x8 acc = {0.f, 0.f, 0.f, 0.f, 0.f, 0.f, 0.f, 0.f};
  for (int kb = 0; kb < Kp; kb += 32) {
    bf16x16 aV = join16(arow + kb, arow + kb + 16);
    bf16x16 bV = join16(brow + kb, brow + kb + 16);
    acc = WMMA_BF16(aV, bV, acc);
  }

  float badd = 0.f;
  if (b0) badd += b0[nB];
  if (b1) badd += b1[nB];
#pragma unroll
  for (int r = 0; r < 8; ++r) {
    const int mOut = blockIdx.y * 16 + r + (half << 3);
    C[(size_t)mOut * ldc + nB] = acc[r] + badd;
  }
}

// ---------------------------------------------------------------------------
// Persistent recurrent LSTM: one workgroup loops over S timesteps.
// h lives in LDS as bf16 (batch padded 8->16); per step g = h @ Whh^T via
// WMMA (A from LDS b128 loads, B from packed bf16 Whh), gates on VALU.
// ---------------------------------------------------------------------------
template <int HH>
__global__ void __launch_bounds__(256)
lstm_recurrent(const float* __restrict__ xg,      // [B, S, 4*HH]
               const __bf16* __restrict__ WhhP,   // [4*HH, HH] packed bf16
               float* __restrict__ hs,            // [B, S, HH]
               float* __restrict__ gbuf,          // [16, 4*HH] global scratch
               int BATCH, int S) {
  constexpr int N = 4 * HH;
  __shared__ __bf16 h_sm[16 * HH];
  __shared__ float  c_sm[8 * HH];

  const int tid  = threadIdx.x;
  const int lane = tid & 31;
  const int wave = tid >> 5;
  const int nWav = blockDim.x >> 5;
  const int half = lane >> 4;
  const int l16  = lane & 15;

  for (int i = tid; i < 16 * HH; i += blockDim.x) h_sm[i] = (__bf16)0.f;
  for (int i = tid; i < 8 * HH;  i += blockDim.x) c_sm[i] = 0.f;
  __syncthreads();

  const __bf16* arow = h_sm + l16 * HH + (half << 3);

  for (int t = 0; t < S; ++t) {
    // ---- g = h @ Whh^T  (M=16 padded, N=4*HH, K=HH) ----
    for (int nt = wave; nt < N / 16; nt += nWav) {
      const int nB = nt * 16 + l16;
      const __bf16* brow = WhhP + (size_t)nB * HH + (half << 3);
      f32x8 acc = {0.f, 0.f, 0.f, 0.f, 0.f, 0.f, 0.f, 0.f};
      for (int kb = 0; kb < HH; kb += 32) {
        bf16x16 aV = join16(arow + kb, arow + kb + 16);
        bf16x16 bV = join16(brow + kb, brow + kb + 16);
        acc = WMMA_BF16(aV, bV, acc);
      }
#pragma unroll
      for (int r = 0; r < 8; ++r)
        gbuf[(r + (half << 3)) * N + nB] = acc[r];
    }
    __threadfence_block();
    __syncthreads();

    // ---- gates (i,f,g,o), state update ----
    for (int idx = tid; idx < BATCH * HH; idx += blockDim.x) {
      const int m = idx / HH, j = idx % HH;
      const float* xr = xg + ((size_t)m * S + t) * N;
      const float gi = gbuf[m * N + j]          + xr[j];
      const float gf = gbuf[m * N + HH + j]     + xr[HH + j];
      const float gg = gbuf[m * N + 2 * HH + j] + xr[2 * HH + j];
      const float go = gbuf[m * N + 3 * HH + j] + xr[3 * HH + j];
      const float si = 1.f / (1.f + expf(-gi));
      const float sf = 1.f / (1.f + expf(-gf));
      const float so = 1.f / (1.f + expf(-go));
      const float cn = sf * c_sm[idx] + si * tanhf(gg);
      const float hn = so * tanhf(cn);
      c_sm[idx] = cn;
      h_sm[m * HH + j] = (__bf16)hn;
      hs[((size_t)m * S + t) * HH + j] = hn;
    }
    __threadfence_block();
    __syncthreads();
  }
}

// ---------------------------------------------------------------------------
// Additive cross attention: one block per (b, w).
// ---------------------------------------------------------------------------
__global__ void __launch_bounds__(256)
xattn_kernel(const float* __restrict__ T, const float* __restrict__ Ssrc,
             const float* __restrict__ w2, const float* __restrict__ src,
             const int* __restrict__ lengths,
             float* __restrict__ attn_w,   // [B,S,S]
             float* __restrict__ outp,     // [(b*S+w)*ld_out + h]
             int S, int Hd, int ld_out) {
  __shared__ float t_sm[384], w2_sm[384], sc[160];
  const int b = blockIdx.y, w = blockIdx.x, tid = threadIdx.x;
  const float* trow = T + ((size_t)b * S + w) * Hd;
  for (int d = tid; d < Hd; d += blockDim.x) { t_sm[d] = trow[d]; w2_sm[d] = w2[d]; }
  __syncthreads();

  const int len = lengths[b];
  for (int v = tid; v < S; v += blockDim.x) {
    const float* srow = Ssrc + ((size_t)b * S + v) * Hd;
    float acc = 0.f;
    for (int d = 0; d < Hd; ++d) acc += w2_sm[d] * tanhf(t_sm[d] + srow[d]);
    sc[v] = (v < len) ? acc : 0.f;   // multiplicative mask
  }
  __syncthreads();

  float mx = -1e30f;
  for (int v = 0; v < S; ++v) mx = fmaxf(mx, sc[v]);
  float sum = 0.f;
  for (int v = 0; v < S; ++v) sum += expf(sc[v] - mx);
  const float inv = 1.f / sum;
  __syncthreads();

  for (int v = tid; v < S; v += blockDim.x) {
    const float wt = expf(sc[v] - mx) * inv;
    sc[v] = wt;
    attn_w[((size_t)b * S + w) * S + v] = wt;
  }
  __syncthreads();

  for (int h = tid; h < Hd; h += blockDim.x) {
    float acc = 0.f;
    for (int v = 0; v < S; ++v) acc += sc[v] * src[((size_t)b * S + v) * Hd + h];
    outp[((size_t)b * S + w) * ld_out + h] = acc;
  }
}

// ---------------------------------------------------------------------------
// Soft attention over time: weights = softmax(reps @ mp_w + mp_b, axis=S)
// ---------------------------------------------------------------------------
__global__ void __launch_bounds__(256)
softattn_kernel(const float* __restrict__ reps,
                const float* __restrict__ mp_w, const float* __restrict__ mp_b,
                float* __restrict__ soft_w, float* __restrict__ comp,
                int S, int H) {
  __shared__ float sc[160], wsm[128];
  const int b = blockIdx.x, tid = threadIdx.x;
  for (int h = tid; h < H; h += blockDim.x) wsm[h] = mp_w[h];
  __syncthreads();
  for (int s = tid; s < S; s += blockDim.x) {
    const float* row = reps + ((size_t)b * S + s) * H;
    float acc = mp_b[0];
    for (int h = 0; h < H; ++h) acc += row[h] * wsm[h];
    sc[s] = acc;
  }
  __syncthreads();
  float mx = -1e30f;
  for (int s = 0; s < S; ++s) mx = fmaxf(mx, sc[s]);
  float sum = 0.f;
  for (int s = 0; s < S; ++s) sum += expf(sc[s] - mx);
  const float inv = 1.f / sum;
  __syncthreads();
  for (int s = tid; s < S; s += blockDim.x) {
    const float wt = expf(sc[s] - mx) * inv;
    sc[s] = wt;
    soft_w[(size_t)b * S + s] = wt;
  }
  __syncthreads();
  for (int h = tid; h < H; h += blockDim.x) {
    float acc = 0.f;
    for (int s = 0; s < S; ++s) acc += sc[s] * reps[((size_t)b * S + s) * H + h];
    comp[(size_t)b * H + h] = acc;
  }
}

// Copy contiguous [rows,cols] into strided destination slice.
__global__ void copy_slice(const float* __restrict__ src, float* __restrict__ dst,
                           int rows, int cols, int ld_dst) {
  const int i = blockIdx.x * blockDim.x + threadIdx.x;
  if (i < rows * cols) {
    const int r = i / cols, c = i % cols;
    dst[(size_t)r * ld_dst + c] = src[i];
  }
}

// ---------------------------------------------------------------------------
// Head: feat = [c_attn[b,len-1,:384], v_comp, a_comp]; relu(feat@fc1^T+b)@fc2^T+b
// ---------------------------------------------------------------------------
__global__ void __launch_bounds__(128)
head_kernel(const float* __restrict__ c_attn,
            const float* __restrict__ v_comp, const float* __restrict__ a_comp,
            const int* __restrict__ lengths,
            const float* __restrict__ fc1_W, const float* __restrict__ fc1_b,
            const float* __restrict__ fc2_W, const float* __restrict__ fc2_b,
            float* __restrict__ out, int S) {
  __shared__ float red[128];
  const int b = blockIdx.x, h = threadIdx.x;
  const int len = lengths[b];
  const float* crow = c_attn + ((size_t)b * S + (len - 1)) * 384;
  float acc = fc1_b[h];
  for (int k = 0; k < 384; ++k) acc += crow[k] * fc1_W[h * 640 + k];
  for (int k = 0; k < 128; ++k) acc += v_comp[b * 128 + k] * fc1_W[h * 640 + 384 + k];
  for (int k = 0; k < 128; ++k) acc += a_comp[b * 128 + k] * fc1_W[h * 640 + 512 + k];
  red[h] = fmaxf(acc, 0.f) * fc2_W[h];
  __syncthreads();
  for (int off = 64; off > 0; off >>= 1) {
    if (h < off) red[h] += red[h + off];
    __syncthreads();
  }
  if (h == 0) out[b] = red[0] + fc2_b[0];
}

// ---------------------------------------------------------------------------
extern "C" void kernel_launch(void* const* d_in, const int* in_sizes, int n_in,
                              void* d_out, int out_size, void* d_ws, size_t ws_size,
                              hipStream_t stream) {
  (void)in_sizes; (void)n_in; (void)out_size; (void)ws_size;
  constexpr int B = 8, S = 160, H = 128, H3 = 384;
  constexpr int BS = B * S;

  const float* w       = (const float*)d_in[0];
  const float* v       = (const float*)d_in[1];
  const float* a       = (const float*)d_in[2];
  const int*   lengths = (const int*)  d_in[3];
  const float* w_Wih = (const float*)d_in[4],  *w_Whh = (const float*)d_in[5];
  const float* w_bih = (const float*)d_in[6],  *w_bhh = (const float*)d_in[7];
  const float* v_Wih = (const float*)d_in[8],  *v_Whh = (const float*)d_in[9];
  const float* v_bih = (const float*)d_in[10], *v_bhh = (const float*)d_in[11];
  const float* a_Wih = (const float*)d_in[12], *a_Whh = (const float*)d_in[13];
  const float* a_bih = (const float*)d_in[14], *a_bhh = (const float*)d_in[15];
  const float* c_Wih = (const float*)d_in[16], *c_Whh = (const float*)d_in[17];
  const float* c_bih = (const float*)d_in[18], *c_bhh = (const float*)d_in[19];
  const float* vw_W1 = (const float*)d_in[20], *vw_b1 = (const float*)d_in[21], *vw_w2 = (const float*)d_in[22];
  const float* aw_W1 = (const float*)d_in[23], *aw_b1 = (const float*)d_in[24], *aw_w2 = (const float*)d_in[25];
  const float* sa_W1 = (const float*)d_in[26], *sa_b1 = (const float*)d_in[27], *sa_w2 = (const float*)d_in[28];
  const float* mp_w  = (const float*)d_in[29], *mp_b  = (const float*)d_in[30];
  const float* fc1_W = (const float*)d_in[31], *fc1_b = (const float*)d_in[32];
  const float* fc2_W = (const float*)d_in[33], *fc2_b = (const float*)d_in[34];

  float* dout = (float*)d_out;
  float* out_scalar = dout;
  float* v_attn     = dout + 8;
  float* v_soft     = dout + 8 + BS * S;
  float* a_attn     = v_soft + BS;
  float* a_soft     = a_attn + BS * S;

  float* ws = (float*)d_ws;
  size_t o = 0;
  auto alloc = [&](size_t n) { float* p = ws + o; o += n; return p; };
  float* w_hs  = alloc((size_t)BS * H);
  float* v_hs  = alloc((size_t)BS * H);
  float* a_hs  = alloc((size_t)BS * H);
  float* cat   = alloc((size_t)BS * H3);
  float* c_hs  = alloc((size_t)BS * H3);
  float* xg    = alloc((size_t)BS * 4 * H3);
  float* gbuf  = alloc((size_t)16 * 4 * H3);
  float* tb    = alloc((size_t)BS * H3);
  float* sb    = alloc((size_t)BS * H3);
  float* catt  = alloc((size_t)BS * H3);
  float* satt  = alloc((size_t)BS * S);
  float* vcomp = alloc((size_t)B * H);
  float* acomp = alloc((size_t)B * H);
  // bf16 packed-operand scratch (reused between consecutive GEMM/LSTM ops)
  __bf16* xp = (__bf16*)(ws + o);                 // up to BS x 384 bf16
  __bf16* wp = xp + (size_t)BS * H3;              // up to 1536 x 384 bf16

  auto pack = [&](const float* src, int ld, __bf16* dst, int rows, int K, int Kp) {
    const int n = rows * Kp;
    pack_bf16<<<(n + 255) / 256, 256, 0, stream>>>(src, ld, dst, rows, K, Kp);
  };
  auto gemm = [&](const __bf16* Xp, const __bf16* Wp, int Kp,
                  const float* b0, const float* b1,
                  float* C, int ldc, int M, int N) {
    gemm_bf16_wmma<<<dim3(N / 16, M / 16), 32, 0, stream>>>(Xp, Wp, Kp, b0, b1,
                                                            C, ldc, N);
  };

  // --- w LSTM (K=300 -> Kp=320) ---
  pack(w, 300, xp, BS, 300, 320);
  pack(w_Wih, 300, wp, 4 * H, 300, 320);
  gemm(xp, wp, 320, w_bih, w_bhh, xg, 4 * H, BS, 4 * H);
  pack(w_Whh, H, wp, 4 * H, H, H);
  lstm_recurrent<128><<<1, 256, 0, stream>>>(xg, wp, w_hs, gbuf, B, S);

  // --- v LSTM (K=35 -> Kp=64) ---
  pack(v, 35, xp, BS, 35, 64);
  pack(v_Wih, 35, wp, 4 * H, 35, 64);
  gemm(xp, wp, 64, v_bih, v_bhh, xg, 4 * H, BS, 4 * H);
  pack(v_Whh, H, wp, 4 * H, H, H);
  lstm_recurrent<128><<<1, 256, 0, stream>>>(xg, wp, v_hs, gbuf, B, S);

  // --- a LSTM (K=74 -> Kp=96) ---
  pack(a, 74, xp, BS, 74, 96);
  pack(a_Wih, 74, wp, 4 * H, 74, 96);
  gemm(xp, wp, 96, a_bih, a_bhh, xg, 4 * H, BS, 4 * H);
  pack(a_Whh, H, wp, 4 * H, H, H);
  lstm_recurrent<128><<<1, 256, 0, stream>>>(xg, wp, a_hs, gbuf, B, S);

  // cat[:, 0:128] = w_hs
  copy_slice<<<(BS * H + 255) / 256, 256, 0, stream>>>(w_hs, cat, BS, H, H3);

  // --- vw cross attention (src=v_hs, tgt=w_hs); out -> cat[:,128:256] ---
  pack(v_hs, H, xp, BS, H, H);
  pack(vw_W1, 2 * H, wp, H, H, H);
  gemm(xp, wp, H, nullptr, nullptr, sb, H, BS, H);
  pack(w_hs, H, xp, BS, H, H);
  pack(vw_W1 + H, 2 * H, wp, H, H, H);
  gemm(xp, wp, H, vw_b1, nullptr, tb, H, BS, H);
  xattn_kernel<<<dim3(S, B), 256, 0, stream>>>(tb, sb, vw_w2, v_hs, lengths,
                                               v_attn, cat + H, S, H, H3);

  // --- aw cross attention; out -> cat[:,256:384] ---
  pack(a_hs, H, xp, BS, H, H);
  pack(aw_W1, 2 * H, wp, H, H, H);
  gemm(xp, wp, H, nullptr, nullptr, sb, H, BS, H);
  pack(w_hs, H, xp, BS, H, H);
  pack(aw_W1 + H, 2 * H, wp, H, H, H);
  gemm(xp, wp, H, aw_b1, nullptr, tb, H, BS, H);
  xattn_kernel<<<dim3(S, B), 256, 0, stream>>>(tb, sb, aw_w2, a_hs, lengths,
                                               a_attn, cat + 2 * H, S, H, H3);

  // --- c LSTM on concatenated features (K=384) ---
  pack(cat, H3, xp, BS, H3, H3);
  pack(c_Wih, H3, wp, 4 * H3, H3, H3);
  gemm(xp, wp, H3, c_bih, c_bhh, xg, 4 * H3, BS, 4 * H3);
  pack(c_Whh, H3, wp, 4 * H3, H3, H3);
  lstm_recurrent<384><<<1, 256, 0, stream>>>(xg, wp, c_hs, gbuf, B, S);

  // --- sa self attention on c_hs ---
  pack(c_hs, H3, xp, BS, H3, H3);
  pack(sa_W1, 2 * H3, wp, H3, H3, H3);
  gemm(xp, wp, H3, nullptr, nullptr, sb, H3, BS, H3);
  pack(sa_W1 + H3, 2 * H3, wp, H3, H3, H3);
  gemm(xp, wp, H3, sa_b1, nullptr, tb, H3, BS, H3);
  xattn_kernel<<<dim3(S, B), 256, 0, stream>>>(tb, sb, sa_w2, c_hs, lengths,
                                               satt, catt, S, H3, H3);

  // --- soft attentions ---
  softattn_kernel<<<B, 256, 0, stream>>>(v_hs, mp_w, mp_b, v_soft, vcomp, S, H);
  softattn_kernel<<<B, 256, 0, stream>>>(a_hs, mp_w, mp_b, a_soft, acomp, S, H);

  // --- head ---
  head_kernel<<<B, 128, 0, stream>>>(catt, vcomp, acomp, lengths,
                                     fc1_W, fc1_b, fc2_W, fc2_b, out_scalar, S);
}
